// SpatialAttention_81269371175321
// MI455X (gfx1250) — matrix operand order repaired and checked by reading, not compiled, not run on hardware
//
#include <hip/hip_runtime.h>
#include <cstdint>

typedef __attribute__((ext_vector_type(16))) _Float16 v16h;
typedef __attribute__((ext_vector_type(8)))  _Float16 v8h;
typedef __attribute__((ext_vector_type(4)))  _Float16 v4h;
typedef __attribute__((ext_vector_type(8)))  float    v8f;
typedef __attribute__((ext_vector_type(4)))  int      v4i;

#define FRAMES 16
#define P2     256
#define HEADS  8
#define HD     64
#define CDIM   512
#define BATCH  8
#define MTOT   (BATCH * FRAMES * P2)   // 32768 tokens

static __device__ __forceinline__ v16h cat8(v8h a, v8h b) {
  return __builtin_shufflevector(a, b, 0,1,2,3,4,5,6,7,8,9,10,11,12,13,14,15);
}
static __device__ __forceinline__ v8f zero8() {
  v8f v = {0.f,0.f,0.f,0.f,0.f,0.f,0.f,0.f};
  return v;
}

// ---------------------------------------------------------------------------
// gfx1250 async global->LDS copy (ASYNCcnt-tracked), with sync fallback.
// Builtin prototype (per hipcc diagnostic): (v4i*, v4i*, imm int, imm int),
// generic pointers, order (global src, LDS dst, offset, cpol).
// ---------------------------------------------------------------------------
#if defined(__has_builtin)
#if __has_builtin(__builtin_amdgcn_global_load_async_to_lds_b128) && \
    __has_builtin(__builtin_amdgcn_s_wait_asynccnt)
#define HAVE_ASYNC_LDS 1
#endif
#endif

static __device__ __forceinline__ void cp16_to_lds(const _Float16* g, _Float16* l) {
#ifdef HAVE_ASYNC_LDS
  __builtin_amdgcn_global_load_async_to_lds_b128((v4i*)g, (v4i*)l, 0, 0);
#else
  *(v8h*)l = *(const v8h*)g;
#endif
}
static __device__ __forceinline__ void cp_wait() {
#ifdef HAVE_ASYNC_LDS
  __builtin_amdgcn_s_wait_asynccnt(0);
#endif
}

// ---------------------------------------------------------------------------
// fp32 -> f16 conversion, vectorized x4 (for x: 16.7M elements)
// ---------------------------------------------------------------------------
__global__ void cvt_f32_f16_v4(const float4* __restrict__ in,
                               v4h* __restrict__ out, int n4) {
  int i = blockIdx.x * 256 + threadIdx.x;
  if (i < n4) {
    float4 f = in[i];
    v4h h = {(_Float16)f.x, (_Float16)f.y, (_Float16)f.z, (_Float16)f.w};
    out[i] = h;
  }
}

// ---------------------------------------------------------------------------
// fp32 -> f16 with transpose (512x512 weights): Wt[n][k] = W[k][n]
// ---------------------------------------------------------------------------
__global__ void cvt_f32_f16_t(const float* __restrict__ in,
                              _Float16* __restrict__ out) {
  int i = blockIdx.x * 256 + threadIdx.x;     // i = k*512 + n
  int k = i >> 9, n = i & 511;
  out[((size_t)n << 9) | (size_t)k] = (_Float16)in[i];
}

// ---------------------------------------------------------------------------
// Tiled WMMA GEMM: Y[32768,512] = X[M,512] * W, W given transposed Wt[n][k].
//  MODE 0: f16 natural [m][n]            (scaled by oscale)
//  MODE 1: f16 q/k layout [bth][p][d]    (scaled by oscale; q uses 0.125)
//  MODE 2: f16 v layout   [bth][d][p]
//  MODE 3: fp32 natural + bias (final output)
// Workgroup: 128 threads = 4 waves; workgroup tile 64x128, wave tile 32x64.
// Double-buffered LDS; stage i+1 copied (async on gfx1250) during stage i's
// 16 WMMAs per wave; one barrier per 64-K stage.
// ---------------------------------------------------------------------------
#define GSTRIDE 72   // LDS row stride in halves (64 + 8 pad, 16B multiple)
#define KSTAGES (CDIM / 64)

template <int MODE>
__global__ __launch_bounds__(128) void gemm16_kernel(
    const _Float16* __restrict__ X,
    const _Float16* __restrict__ Wt,   // [n][k]
    _Float16* __restrict__ Yh,
    float* __restrict__ Yf,
    const float* __restrict__ bias,
    float oscale)
{
  __shared__ _Float16 Alds[2][64 * GSTRIDE];    // [buf][row m][k0..63]
  __shared__ _Float16 Blds[2][128 * GSTRIDE];   // [buf][col n][k0..63]

  const int tid  = threadIdx.x;
  const int lane = tid & 31;
  const int lw   = lane & 15;
  const int hi   = lane >> 4;
  const int wave = tid >> 5;
  const int wr   = wave >> 1;          // wave row 0..1 (32 rows each)
  const int wc   = wave & 1;           // wave col 0..1 (64 cols each)
  const int m0   = blockIdx.x * 64;
  const int n0   = blockIdx.y * 128;

  v8f acc[2][4];
#pragma unroll
  for (int r = 0; r < 2; ++r)
#pragma unroll
    for (int c = 0; c < 4; ++c) acc[r][c] = zero8();

  const int arow  = tid >> 1;          // 0..63
  const int apart = (tid & 1) * 32;    // halves 0..31 or 32..63

  auto stage = [&](int kk, int buf) {
    // A tile: 64 rows x 64 halves (4 x 16B per thread)
    const _Float16* asrc = X + (size_t)(m0 + arow) * CDIM + kk + apart;
    _Float16* adst = Alds[buf] + arow * GSTRIDE + apart;
#pragma unroll
    for (int c = 0; c < 4; ++c) cp16_to_lds(asrc + c * 8, adst + c * 8);
    // B tile: 128 cols x 64 halves (8 x 16B per thread)
    const _Float16* bsrc = Wt + (size_t)(n0 + tid) * CDIM + kk;
    _Float16* bdst = Blds[buf] + tid * GSTRIDE;
#pragma unroll
    for (int c = 0; c < 8; ++c) cp16_to_lds(bsrc + c * 8, bdst + c * 8);
  };

  stage(0, 0);                          // prologue
  for (int it = 0; it < KSTAGES; ++it) {
    const int buf = it & 1;
    cp_wait();                          // own async copies done
    __syncthreads();                    // everyone's copies visible
    if (it + 1 < KSTAGES) stage((it + 1) * 64, buf ^ 1);   // overlap w/ compute

    // --- two K=32 WMMA steps on this stage
    // A: lane row = M (lane%16), K = 8*hi + {0..7} U 16+8*hi + {0..7}
    // B: lane col = N (lane%16), K = 8*hi + {0..7} U 16+8*hi + {0..7}
#pragma unroll
    for (int s = 0; s < 2; ++s) {
      v16h af[2], bf[4];
#pragma unroll
      for (int rt = 0; rt < 2; ++rt) {
        const _Float16* ap =
            Alds[buf] + (wr * 32 + rt * 16 + lw) * GSTRIDE + s * 32 + 8 * hi;
        af[rt] = cat8(*(const v8h*)ap, *(const v8h*)(ap + 16));
      }
#pragma unroll
      for (int ct = 0; ct < 4; ++ct) {
        const _Float16* bp =
            Blds[buf] + (wc * 64 + ct * 16 + lw) * GSTRIDE + s * 32 + 8 * hi;
        bf[ct] = cat8(*(const v8h*)bp, *(const v8h*)(bp + 16));
      }
#pragma unroll
      for (int rt = 0; rt < 2; ++rt)
#pragma unroll
        for (int ct = 0; ct < 4; ++ct)
          acc[rt][ct] = __builtin_amdgcn_wmma_f32_16x16x32_f16(
              false, af[rt], false, bf[ct], (short)0, acc[rt][ct], false, false);
    }
  }

  // --- store (MODE compile-time; dead branches fold)
#pragma unroll
  for (int rt = 0; rt < 2; ++rt) {
#pragma unroll
    for (int ct = 0; ct < 4; ++ct) {
      const int gn = n0 + wc * 64 + ct * 16 + lw;
      const int h  = gn >> 6, d = gn & 63;
#pragma unroll
      for (int j = 0; j < 8; ++j) {
        const int gm = m0 + wr * 32 + rt * 16 + 8 * hi + j;
        const float val = acc[rt][ct][j];
        if (MODE == 0) {
          Yh[(size_t)gm * CDIM + gn] = (_Float16)(val * oscale);
        } else if (MODE == 1) {
          const int bt = gm >> 8, p = gm & 255;
          Yh[((size_t)(bt * HEADS + h) * P2 + p) * HD + d] = (_Float16)(val * oscale);
        } else if (MODE == 2) {
          const int bt = gm >> 8, p = gm & 255;
          Yh[((size_t)(bt * HEADS + h) * HD + d) * P2 + p] = (_Float16)(val * oscale);
        } else {
          Yf[(size_t)gm * CDIM + gn] = val + bias[gn];
        }
      }
    }
  }
}

// ---------------------------------------------------------------------------
// Attention: one workgroup per (b, frame, head). 256 keys, head_dim 64.
// Q is pre-scaled by 1/sqrt(64) at projection time, so no score scaling here.
// Computes S^T = K*Q^T tiles (softmax rows along VGPR axis), 2-pass softmax,
// O^T = V^T * P^T via WMMA. Output f16 natural [m][512].
// ---------------------------------------------------------------------------
#define KSTRIDE 72    // padded LDS stride for K rows (halves)
#define VSTRIDE 264   // padded LDS stride for V^T rows (halves)
#define ATTN_LDS_BYTES ((P2 * KSTRIDE + HD * VSTRIDE) * 2)

__global__ __launch_bounds__(256) void attn_kernel(
    const _Float16* __restrict__ Qg,   // [BTH][256][64]  (pre-scaled)
    const _Float16* __restrict__ Kg,   // [BTH][256][64]
    const _Float16* __restrict__ Vg,   // [BTH][64][256]  (transposed)
    _Float16* __restrict__ Og)         // [32768][512]
{
  extern __shared__ char smem[];
  _Float16* Klds = (_Float16*)smem;                       // [256][KSTRIDE]
  _Float16* Vt   = (_Float16*)(smem + P2 * KSTRIDE * 2);  // [64][VSTRIDE]

  const int pid  = blockIdx.x;           // (b*T+t)*H + h
  const int h    = pid & 7;
  const int bt   = pid >> 3;
  const size_t base = (size_t)pid * P2 * HD;
  const int tid  = threadIdx.x;
  const int lane = tid & 31;
  const int lw   = lane & 15;
  const int hi   = lane >> 4;
  const int wave = tid >> 5;
  const int q0   = wave * 32;            // 8 waves x 32 queries

  // --- stage K (natural rows) into padded LDS
  {
    const _Float16* src = Kg + base + (size_t)tid * HD;
    _Float16* dst = Klds + tid * KSTRIDE;
#pragma unroll
    for (int c = 0; c < 8; ++c) cp16_to_lds(src + c * 8, dst + c * 8);
  }
  // --- stage V^T (already transposed in global) into padded LDS
  {
    int row = tid >> 2, cq = (tid & 3) * 64;
    const _Float16* src = Vg + base + (size_t)row * P2 + cq;
    _Float16* dst = Vt + row * VSTRIDE + cq;
#pragma unroll
    for (int c = 0; c < 8; ++c) cp16_to_lds(src + c * 8, dst + c * 8);
  }

  // --- Q B-fragments held in registers: col = query(lane%16), K(d)=8*hi+{0..7} U 16+8*hi+{0..7}
  v16h qf[2][2];
#pragma unroll
  for (int qt = 0; qt < 2; ++qt)
#pragma unroll
    for (int s = 0; s < 2; ++s) {
      const _Float16* qp =
          Qg + base + (size_t)(q0 + qt * 16 + lw) * HD + s * 32 + 8 * hi;
      qf[qt][s] = cat8(*(const v8h*)qp, *(const v8h*)(qp + 16));
    }
  cp_wait();
  __syncthreads();

  // ---- pass 1: row max of scores
  float mx[2] = {-1e30f, -1e30f};
  for (int kb = 0; kb < 16; ++kb) {
    v16h kf[2];
#pragma unroll
    for (int s = 0; s < 2; ++s) {
      const _Float16* kp = Klds + (kb * 16 + lw) * KSTRIDE + s * 32 + 8 * hi;
      kf[s] = cat8(*(const v8h*)kp, *(const v8h*)(kp + 16));
    }
#pragma unroll
    for (int qt = 0; qt < 2; ++qt) {
      v8f c = zero8();
      c = __builtin_amdgcn_wmma_f32_16x16x32_f16(false, kf[0], false, qf[qt][0], (short)0, c, false, false);
      c = __builtin_amdgcn_wmma_f32_16x16x32_f16(false, kf[1], false, qf[qt][1], (short)0, c, false, false);
#pragma unroll
      for (int j = 0; j < 8; ++j) mx[qt] = fmaxf(mx[qt], c[j]);
    }
  }
#pragma unroll
  for (int qt = 0; qt < 2; ++qt)
    mx[qt] = fmaxf(mx[qt], __shfl_xor(mx[qt], 16, 32));

  // ---- pass 2: exp, sum, and O^T accumulation
  float sm[2] = {0.f, 0.f};
  v8f o[4][2];
#pragma unroll
  for (int dt = 0; dt < 4; ++dt)
#pragma unroll
    for (int qt = 0; qt < 2; ++qt) o[dt][qt] = zero8();

  for (int sb = 0; sb < 8; ++sb) {     // 32-key superblocks
    v16h kf0[2], kf1[2], vf[4];
#pragma unroll
    for (int s = 0; s < 2; ++s) {
      const _Float16* kp0 = Klds + ((2 * sb) * 16 + lw) * KSTRIDE + s * 32 + 8 * hi;
      const _Float16* kp1 = Klds + ((2 * sb + 1) * 16 + lw) * KSTRIDE + s * 32 + 8 * hi;
      kf0[s] = cat8(*(const v8h*)kp0, *(const v8h*)(kp0 + 16));
      kf1[s] = cat8(*(const v8h*)kp1, *(const v8h*)(kp1 + 16));
    }
#pragma unroll
    for (int dt = 0; dt < 4; ++dt) {
      const _Float16* vp = Vt + (dt * 16 + lw) * VSTRIDE + sb * 32 + 16 * hi;
      vf[dt] = cat8(*(const v8h*)vp, *(const v8h*)(vp + 8));
    }
#pragma unroll
    for (int qt = 0; qt < 2; ++qt) {
      v8f c0 = zero8(), c1 = zero8();
      c0 = __builtin_amdgcn_wmma_f32_16x16x32_f16(false, kf0[0], false, qf[qt][0], (short)0, c0, false, false);
      c0 = __builtin_amdgcn_wmma_f32_16x16x32_f16(false, kf0[1], false, qf[qt][1], (short)0, c0, false, false);
      c1 = __builtin_amdgcn_wmma_f32_16x16x32_f16(false, kf1[0], false, qf[qt][0], (short)0, c1, false, false);
      c1 = __builtin_amdgcn_wmma_f32_16x16x32_f16(false, kf1[1], false, qf[qt][1], (short)0, c1, false, false);
      float p0[8], p1[8];
#pragma unroll
      for (int j = 0; j < 8; ++j) {
        p0[j] = __expf(c0[j] - mx[qt]);
        p1[j] = __expf(c1[j] - mx[qt]);
        sm[qt] += p0[j] + p1[j];
      }
      // pack exp(S^T) tiles into a B-fragment (keys = K-dim, col = query):
      // lane<16 needs keys 0..15 (block kb0), lane>=16 keys 16..31 (kb1)
      v16h pf;
#pragma unroll
      for (int j = 0; j < 8; ++j) {
        float e0 = __shfl_xor(p0[j], 16, 32);
        float e1 = __shfl_xor(p1[j], 16, 32);
        float lo = hi ? e1 : p0[j];
        float hh = hi ? p1[j] : e0;
        pf[j]     = (_Float16)lo;
        pf[j + 8] = (_Float16)hh;
      }
#pragma unroll
      for (int dt = 0; dt < 4; ++dt)
        o[dt][qt] = __builtin_amdgcn_wmma_f32_16x16x32_f16(
            false, vf[dt], false, pf, (short)0, o[dt][qt], false, false);
    }
  }

  // ---- normalize and store (f16 natural [m][512])
#pragma unroll
  for (int qt = 0; qt < 2; ++qt) {
    float s = sm[qt] + __shfl_xor(sm[qt], 16, 32);
    float inv = 1.0f / s;
    const int q = q0 + qt * 16 + lw;
    const size_t mtok = (size_t)bt * P2 + q;
#pragma unroll
    for (int dt = 0; dt < 4; ++dt) {
      v8h st;
#pragma unroll
      for (int j = 0; j < 8; ++j) st[j] = (_Float16)(o[dt][qt][j] * inv);
      *(v8h*)(Og + mtok * CDIM + h * HD + dt * 16 + 8 * hi) = st;
    }
  }
}

// ---------------------------------------------------------------------------
extern "C" void kernel_launch(void* const* d_in, const int* in_sizes, int n_in,
                              void* d_out, int out_size, void* d_ws, size_t ws_size,
                              hipStream_t stream) {
  (void)in_sizes; (void)n_in; (void)out_size; (void)ws_size;
  const float* x  = (const float*)d_in[0];
  const float* Wq = (const float*)d_in[1];
  const float* Wk = (const float*)d_in[2];
  const float* Wv = (const float*)d_in[3];
  const float* Wo = (const float*)d_in[4];
  const float* bo = (const float*)d_in[5];

  const size_t CW = (size_t)CDIM * CDIM;   // 262144
  const size_t XE = (size_t)MTOT * CDIM;   // 16777216
  _Float16* wqh = (_Float16*)d_ws;         // weights stored transposed [n][k]
  _Float16* wkh = wqh + CW;
  _Float16* wvh = wkh + CW;
  _Float16* woh = wvh + CW;
  _Float16* xh  = woh + CW;     // also reused as attention output buffer
  _Float16* qh  = xh + XE;
  _Float16* kh  = qh + XE;
  _Float16* vh  = kh + XE;

  // fp32 -> f16
  cvt_f32_f16_t<<<(int)(CW / 256), 256, 0, stream>>>(Wq, wqh);
  cvt_f32_f16_t<<<(int)(CW / 256), 256, 0, stream>>>(Wk, wkh);
  cvt_f32_f16_t<<<(int)(CW / 256), 256, 0, stream>>>(Wv, wvh);
  cvt_f32_f16_t<<<(int)(CW / 256), 256, 0, stream>>>(Wo, woh);
  cvt_f32_f16_v4<<<(int)(XE / 4 / 256), 256, 0, stream>>>(
      (const float4*)x, (v4h*)xh, (int)(XE / 4));

  dim3 gdim(MTOT / 64, CDIM / 128);
  // QKV projections (q pre-scaled by 1/sqrt(head_dim))
  gemm16_kernel<1><<<gdim, 128, 0, stream>>>(xh, wqh, qh, nullptr, nullptr, 0.125f);
  gemm16_kernel<1><<<gdim, 128, 0, stream>>>(xh, wkh, kh, nullptr, nullptr, 1.0f);
  gemm16_kernel<2><<<gdim, 128, 0, stream>>>(xh, wvh, vh, nullptr, nullptr, 1.0f);
  // attention per (batch, frame, head); writes into xh (no longer needed)
  attn_kernel<<<BATCH * FRAMES * HEADS, 256, ATTN_LDS_BYTES, stream>>>(qh, kh, vh, xh);
  // output projection + bias, fp32
  gemm16_kernel<3><<<gdim, 128, 0, stream>>>(xh, woh, nullptr, (float*)d_out, bo, 1.0f);
}
